// LinkPredictor_77498389889811
// MI455X (gfx1250) — matrix operand order repaired and checked by reading, not compiled, not run on hardware
//
#include <hip/hip_runtime.h>
#include <hip/hip_bf16.h>
#include <math.h>

typedef __attribute__((ext_vector_type(16))) _Float16 v16h;
typedef __attribute__((ext_vector_type(8)))  _Float16 v8h;
typedef __attribute__((ext_vector_type(8)))  float    v8f;

// ---------------------------------------------------------------------------
// GEMM: C[M,N] = A[M,K] @ W[K,N], f32 in/out, f16 WMMA compute, f32 accum.
// Block = 256 thr (8 waves). blockIdx.y picks a 16-column N-strip; the strip's
// W panel is staged once into LDS (transposed, padded), each wave hoists ALL
// K/32 B fragments into VGPRs, then grid-strides over 16-row M tiles so B is
// loaded from global exactly once per strip. Requires M%16==0, K%32==0,
// N%16==0 (true here: M=50000, K in {128,256}, N in {256,16}).
// ---------------------------------------------------------------------------
template <int K>
__global__ void gemm_wmma_f16_k(const float* __restrict__ A,
                                const float* __restrict__ W,
                                float* __restrict__ C,
                                int M, int N) {
  constexpr int KF = K / 32;       // # K fragments
  constexpr int KP = K + 8;        // padded LDS row length (halves)
  __shared__ alignas(16) _Float16 lw[16 * KP];   // [col][k] transposed panel

  const int col0 = blockIdx.y << 4;

  // cooperative stage: W[k][col0+c] -> lw[c*KP + k]  (f32 -> f16)
  for (int idx = threadIdx.x; idx < K * 16; idx += blockDim.x) {
    int k = idx >> 4, c = idx & 15;
    lw[c * KP + k] = (_Float16)W[(size_t)k * N + col0 + c];
  }
  __syncthreads();

  const int lane = threadIdx.x & 31;
  const int wave = threadIdx.x >> 5;
  const int r    = lane & 15;
  const int hi   = lane >> 4;

  // Hoist B fragments into registers: lane holds col=r, halves 0..7 -> K=kf*32+hi*8+e ; halves 8..15 -> +16
  v16h bfrag[KF];
#pragma unroll
  for (int kf = 0; kf < KF; ++kf) {
    const _Float16* p = &lw[r * KP + kf * 32 + hi * 8];
    v8h b0 = *reinterpret_cast<const v8h*>(p);       // 16B ds_load
    v8h b1 = *reinterpret_cast<const v8h*>(p + 16);  // 16B ds_load
    bfrag[kf] = __builtin_shufflevector(b0, b1, 0, 1, 2, 3, 4, 5, 6, 7,
                                        8, 9, 10, 11, 12, 13, 14, 15);
  }

  const int mtiles     = M >> 4;
  const int wavesTotal = gridDim.x * (blockDim.x >> 5);

  for (int mt = blockIdx.x * (blockDim.x >> 5) + wave; mt < mtiles; mt += wavesTotal) {
    const float* __restrict__ arow = A + (size_t)((mt << 4) + r) * K;
    int mtn = mt + wavesTotal;
    if (mtn < mtiles)   // hint next A tile into cache (global_prefetch_b8)
      __builtin_prefetch(A + (size_t)((mtn << 4) + r) * K, 0, 0);

    v8f acc = {};
#pragma unroll
    for (int kf = 0; kf < KF; ++kf) {
      const int k0 = kf * 32;
      // A fragment: halves 0..7 -> K=k0+hi*8+e ; halves 8..15 -> +16
      const float4* ap0 = reinterpret_cast<const float4*>(arow + k0 + hi * 8);
      const float4* ap1 = reinterpret_cast<const float4*>(arow + k0 + 16 + hi * 8);
      float4 a0 = ap0[0], a1 = ap0[1], a2 = ap1[0], a3 = ap1[1];
      v16h af;
      af[0]=(_Float16)a0.x; af[1]=(_Float16)a0.y; af[2]=(_Float16)a0.z; af[3]=(_Float16)a0.w;
      af[4]=(_Float16)a1.x; af[5]=(_Float16)a1.y; af[6]=(_Float16)a1.z; af[7]=(_Float16)a1.w;
      af[8]=(_Float16)a2.x; af[9]=(_Float16)a2.y; af[10]=(_Float16)a2.z; af[11]=(_Float16)a2.w;
      af[12]=(_Float16)a3.x; af[13]=(_Float16)a3.y; af[14]=(_Float16)a3.z; af[15]=(_Float16)a3.w;

      acc = __builtin_amdgcn_wmma_f32_16x16x32_f16(false, af, false, bfrag[kf],
                                                   (short)0, acc, false, false);
    }
    // D: VGPR i holds row (mt*16 + hi*8 + i), col col0 + r
    float* __restrict__ crow = C + (size_t)((mt << 4) + hi * 8) * N + col0 + r;
#pragma unroll
    for (int i = 0; i < 8; ++i) crow[(size_t)i * N] = acc[i];
  }
}

// ---------------------------------------------------------------------------
// Small helpers
// ---------------------------------------------------------------------------
__global__ void fill_i32(int* p, int v, int n) {
  int i = blockIdx.x * blockDim.x + threadIdx.x;
  if (i < n) p[i] = v;
}

// wd_vec[k] = sum_j Wd[k][j] * ad[j]
__global__ void mat_vec_right(const float* __restrict__ Wd, const float* __restrict__ ad,
                              float* __restrict__ outv, int K, int N) {
  int k = blockIdx.x * blockDim.x + threadIdx.x;
  if (k >= K) return;
  const float* r = Wd + (size_t)k * N;
  float s = 0.f;
  for (int j = 0; j < N; ++j) s += r[j] * ad[j];
  outv[k] = s;
}

// out[i] = dot(X[i,:], v)  -- one wave per row, float4 lanes (K % 16 == 0 here)
__global__ void rows_dot_vec(const float* __restrict__ X, const float* __restrict__ v,
                             float* __restrict__ out, int nrows, int K) {
  int lane = threadIdx.x & 31;
  int row  = blockIdx.x * (blockDim.x >> 5) + (threadIdx.x >> 5);
  if (row >= nrows) return;
  const float4* x4 = reinterpret_cast<const float4*>(X + (size_t)row * K);
  const float4* v4 = reinterpret_cast<const float4*>(v);
  int K4 = K >> 2;
  float s = 0.f;
  for (int k = lane; k < K4; k += 32) {
    float4 a = x4[k], b = v4[k];
    s += a.x * b.x + a.y * b.y + a.z * b.z + a.w * b.w;
  }
#pragma unroll
  for (int o = 16; o; o >>= 1) s += __shfl_down(s, o, 32);
  if (lane == 0) out[row] = s;
}

// ---------------------------------------------------------------------------
// CSR build (once per direction, reused by all 3 layers)
// ---------------------------------------------------------------------------
__global__ void count_edges(const int* __restrict__ dst, int* __restrict__ cnt, int E) {
  int e = blockIdx.x * blockDim.x + threadIdx.x;
  if (e >= E) return;
  atomicAdd(&cnt[dst[e]], 1);
}

// single-block exclusive scan: rs[0..n] from cnt[0..n-1]
__global__ void excl_scan(const int* __restrict__ cnt, int* __restrict__ rs, int n) {
  __shared__ int buf[256];
  __shared__ int carry;
  if (threadIdx.x == 0) carry = 0;
  __syncthreads();
  for (int base = 0; base < n; base += 256) {
    int i = base + (int)threadIdx.x;
    int v = (i < n) ? cnt[i] : 0;
    buf[threadIdx.x] = v;
    __syncthreads();
    for (int o = 1; o < 256; o <<= 1) {
      int t = (threadIdx.x >= (unsigned)o) ? buf[threadIdx.x - o] : 0;
      __syncthreads();
      buf[threadIdx.x] += t;
      __syncthreads();
    }
    int c = carry;
    if (i < n) rs[i] = c + buf[threadIdx.x] - v;
    __syncthreads();
    if (threadIdx.x == 0) carry = c + buf[255];
    __syncthreads();
  }
  if (threadIdx.x == 0) rs[n] = carry;
}

__global__ void csr_fill(const int* __restrict__ dst, int* __restrict__ cursor,
                         int* __restrict__ perm, int E) {
  int e = blockIdx.x * blockDim.x + threadIdx.x;
  if (e >= E) return;
  int p = atomicAdd(&cursor[dst[e]], 1);
  perm[p] = e;
}

// ---------------------------------------------------------------------------
// GAT edge/aggregation kernels (atomic-free hot path)
// ---------------------------------------------------------------------------
__global__ void edge_compute_e(const int* __restrict__ src, const int* __restrict__ dst,
                               const float* __restrict__ a_s, const float* __restrict__ a_d,
                               float* __restrict__ e_buf, int E) {
  int e = blockIdx.x * blockDim.x + threadIdx.x;
  if (e >= E) return;
  float v = a_s[src[e]] + a_d[dst[e]];
  e_buf[e] = v > 0.f ? v : 0.2f * v;          // leaky_relu(0.2)
}

// one wave per destination node: segment softmax + weighted feature gather,
// fused with bias add and optional ReLU. No atomics.
__global__ void gat_aggregate(const int* __restrict__ rs, const int* __restrict__ perm,
                              const int* __restrict__ src, const float* __restrict__ e_buf,
                              const float* __restrict__ hs, const float* __restrict__ bias,
                              float* __restrict__ outp, int ndst, int F, int relu) {
  int lane = threadIdx.x & 31;
  int d = blockIdx.x * (blockDim.x >> 5) + (threadIdx.x >> 5);
  if (d >= ndst) return;
  int s0 = rs[d], s1 = rs[d + 1];

  // segment max
  float m = -INFINITY;
  for (int j = s0 + lane; j < s1; j += 32) m = fmaxf(m, e_buf[perm[j]]);
#pragma unroll
  for (int o = 16; o; o >>= 1) m = fmaxf(m, __shfl_xor(m, o, 32));
  if (!(m >= -3.402823466e38f && m <= 3.402823466e38f)) m = 0.f;  // isfinite fixup

  // segment sum of exp(e - m)
  float den = 0.f;
  for (int j = s0 + lane; j < s1; j += 32) den += __expf(e_buf[perm[j]] - m);
#pragma unroll
  for (int o = 16; o; o >>= 1) den += __shfl_xor(den, o, 32);
  float rden = 1.f / (den + 1e-16f);

  // weighted gather of source rows; lane owns float4 chunks lane and lane+32
  int F4 = F >> 2;
  float4 acc0 = make_float4(0.f, 0.f, 0.f, 0.f);
  float4 acc1 = make_float4(0.f, 0.f, 0.f, 0.f);
  for (int j = s0; j < s1; ++j) {
    int e = perm[j];
    float alpha = __expf(e_buf[e] - m) * rden;
    const float4* hrow = reinterpret_cast<const float4*>(hs + (size_t)src[e] * F);
    if (lane < F4) {
      float4 h = hrow[lane];
      acc0.x += alpha * h.x; acc0.y += alpha * h.y;
      acc0.z += alpha * h.z; acc0.w += alpha * h.w;
    }
    if (lane + 32 < F4) {
      float4 h = hrow[lane + 32];
      acc1.x += alpha * h.x; acc1.y += alpha * h.y;
      acc1.z += alpha * h.z; acc1.w += alpha * h.w;
    }
  }

  const float4* b4 = reinterpret_cast<const float4*>(bias);
  float4* orow = reinterpret_cast<float4*>(outp + (size_t)d * F);
  if (lane < F4) {
    float4 b = b4[lane];
    float4 o = make_float4(acc0.x + b.x, acc0.y + b.y, acc0.z + b.z, acc0.w + b.w);
    if (relu) {
      o.x = fmaxf(o.x, 0.f); o.y = fmaxf(o.y, 0.f);
      o.z = fmaxf(o.z, 0.f); o.w = fmaxf(o.w, 0.f);
    }
    orow[lane] = o;
  }
  if (lane + 32 < F4) {
    float4 b = b4[lane + 32];
    float4 o = make_float4(acc1.x + b.x, acc1.y + b.y, acc1.z + b.z, acc1.w + b.w);
    if (relu) {
      o.x = fmaxf(o.x, 0.f); o.y = fmaxf(o.y, 0.f);
      o.z = fmaxf(o.z, 0.f); o.w = fmaxf(o.w, 0.f);
    }
    orow[lane + 32] = o;
  }
}

__global__ void classify(const int* __restrict__ src, const int* __restrict__ dst,
                         const float* __restrict__ hA, const float* __restrict__ hB,
                         const float* __restrict__ w, const float* __restrict__ bias,
                         float* __restrict__ out, int EL) {
  int e = blockIdx.x * blockDim.x + threadIdx.x;
  if (e >= EL) return;
  const float* xa = hA + (size_t)src[e] * 16;
  const float* xb = hB + (size_t)dst[e] * 16;
  float s = bias[0];
#pragma unroll
  for (int j = 0; j < 16; ++j) s += xa[j] * w[j] + xb[j] * w[16 + j];
  out[e] = 1.f / (1.f + __expf(-s));
}

// ---------------------------------------------------------------------------
// Host orchestration
// ---------------------------------------------------------------------------
static inline int cdiv(int a, int b) { return (a + b - 1) / b; }

static void launch_gemm(const float* A, const float* W, float* C,
                        int M, int K, int N, hipStream_t stream) {
  int ntN = N >> 4;
  int mtiles = M >> 4;
  int tpw = (ntN >= 8) ? 8 : 1;             // target M-tiles per wave
  dim3 grid(cdiv(mtiles, 8 * tpw), ntN);
  if (K == 128) gemm_wmma_f16_k<128><<<grid, 256, 0, stream>>>(A, W, C, M, N);
  else          gemm_wmma_f16_k<256><<<grid, 256, 0, stream>>>(A, W, C, M, N);
}

struct Csr { int *rs, *perm; };
struct Scratch {
  float *hs, *a_s, *a_d, *e_buf, *wdvec;
};

static void build_csr(const int* dst, int ndst, int E, int* rs, int* cursor,
                      int* perm, hipStream_t stream) {
  fill_i32<<<cdiv(ndst, 256), 256, 0, stream>>>(cursor, 0, ndst);   // cursor as counts
  count_edges<<<cdiv(E, 256), 256, 0, stream>>>(dst, cursor, E);
  excl_scan<<<1, 256, 0, stream>>>(cursor, rs, ndst);
  hipMemcpyAsync(cursor, rs, (size_t)ndst * sizeof(int),
                 hipMemcpyDeviceToDevice, stream);
  csr_fill<<<cdiv(E, 256), 256, 0, stream>>>(dst, cursor, perm, E);
}

static void run_conv(const float* xsrc, int nsrc, const float* xdst, int ndst,
                     int fin, int fout,
                     const int* src, const int* dst, int E, const Csr& csr,
                     const float* Ws, const float* Wd, const float* as_,
                     const float* ad_, const float* b,
                     float* newdst, const Scratch& S, int relu, hipStream_t stream) {
  // hs = xsrc @ Ws  (WMMA)
  launch_gemm(xsrc, Ws, S.hs, nsrc, fin, fout, stream);
  // a_s = hs @ as
  rows_dot_vec<<<cdiv(nsrc, 8), 256, 0, stream>>>(S.hs, as_, S.a_s, nsrc, fout);
  // a_d = xdst @ (Wd @ ad)
  mat_vec_right<<<cdiv(fin, 256), 256, 0, stream>>>(Wd, ad_, S.wdvec, fin, fout);
  rows_dot_vec<<<cdiv(ndst, 8), 256, 0, stream>>>(xdst, S.wdvec, S.a_d, ndst, fin);
  // per-edge attention logits
  edge_compute_e<<<cdiv(E, 256), 256, 0, stream>>>(src, dst, S.a_s, S.a_d, S.e_buf, E);
  // atomic-free segment softmax + aggregation + bias + activation
  gat_aggregate<<<cdiv(ndst, 8), 256, 0, stream>>>(csr.rs, csr.perm, src, S.e_buf,
                                                   S.hs, b, newdst, ndst, fout, relu);
}

extern "C" void kernel_launch(void* const* d_in, const int* in_sizes, int n_in,
                              void* d_out, int out_size, void* d_ws, size_t ws_size,
                              hipStream_t stream) {
  const int IN = 128, HID = 256, OUT = 16;
  const float* x_a = (const float*)d_in[0];
  const float* x_b = (const float*)d_in[1];
  const int*   e_ab = (const int*)d_in[2];
  const int*   e_ba = (const int*)d_in[3];
  const int*   eli  = (const int*)d_in[4];
  const float* cls_b = (const float*)d_in[5];
  const float* cls_w = (const float*)d_in[6];

  const int NA = in_sizes[0] / IN;
  const int NB = in_sizes[1] / IN;
  const int E  = in_sizes[2] / 2;
  const int EL = in_sizes[4] / 2;
  const int NMAX = NA > NB ? NA : NB;
  const int FMAX = HID;

  // pytree (sorted-key) param layout: per layer, per {ab,ba}: Wd, Ws, ad, as, b
  auto P = [&](int layer, int dir /*0=ab,1=ba*/, int which) -> const float* {
    return (const float*)d_in[7 + layer * 10 + dir * 5 + which];
  };

  // carve workspace
  float* ws = (float*)d_ws;
  size_t off = 0;
  float* h0A = ws + off; off += (size_t)NMAX * FMAX;
  float* h0B = ws + off; off += (size_t)NMAX * FMAX;
  float* h1A = ws + off; off += (size_t)NMAX * FMAX;
  float* h1B = ws + off; off += (size_t)NMAX * FMAX;
  Scratch S;
  S.hs    = ws + off; off += (size_t)NMAX * FMAX;
  S.a_s   = ws + off; off += NMAX;
  S.a_d   = ws + off; off += NMAX;
  S.e_buf = ws + off; off += E;
  S.wdvec = ws + off; off += 256;
  Csr csr_ab, csr_ba;
  csr_ab.rs   = (int*)(ws + off); off += NMAX + 1;
  csr_ab.perm = (int*)(ws + off); off += E;
  csr_ba.rs   = (int*)(ws + off); off += NMAX + 1;
  csr_ba.perm = (int*)(ws + off); off += E;
  int* cursor = (int*)(ws + off); off += NMAX + 1;
  (void)ws_size; (void)n_in; (void)out_size;

  const int* ab_src = e_ab;      const int* ab_dst = e_ab + E;
  const int* ba_src = e_ba;      const int* ba_dst = e_ba + E;

  // Build destination-indexed CSRs once; reused by all 3 layers.
  build_csr(ab_dst, NB, E, csr_ab.rs, cursor, csr_ab.perm, stream);
  build_csr(ba_dst, NA, E, csr_ba.rs, cursor, csr_ba.perm, stream);

  float* poolA[2] = { h0A, h1A };
  float* poolB[2] = { h0B, h1B };
  const float* curA = x_a;
  const float* curB = x_b;
  int fin = IN;
  for (int L = 0; L < 3; ++L) {
    int fout = (L == 2) ? OUT : HID;
    int relu = (L < 2) ? 1 : 0;
    float* newA = poolA[L & 1];
    float* newB = poolB[L & 1];
    // ab: A -> B (updates B nodes)
    run_conv(curA, NA, curB, NB, fin, fout, ab_src, ab_dst, E, csr_ab,
             P(L,0,1), P(L,0,0), P(L,0,3), P(L,0,2), P(L,0,4),
             newB, S, relu, stream);
    // ba: B -> A (updates A nodes) -- uses OLD curA/curB
    run_conv(curB, NB, curA, NA, fin, fout, ba_src, ba_dst, E, csr_ba,
             P(L,1,1), P(L,1,0), P(L,1,3), P(L,1,2), P(L,1,4),
             newA, S, relu, stream);
    curA = newA; curB = newB; fin = fout;
  }

  classify<<<cdiv(EL, 256), 256, 0, stream>>>(eli, eli + EL, curA, curB,
                                              cls_w, cls_b, (float*)d_out, EL);
}
